// MAGIC_Actor_76321568850479
// MI455X (gfx1250) — matrix operand order repaired and checked by reading, not compiled, not run on hardware
//
#include <hip/hip_runtime.h>
#include <math.h>

// ---------------- problem constants ----------------
#define BATCH   8192
#define NAG     20
#define OBS_    100
#define OBSP    128     // padded K for obs encoder
#define HDIM    256
#define G4H     1024    // 4*H
#define HEADS_  4
#define GH_     64
#define ACT_    12
#define BN      (BATCH * NAG)          // 163840
#define LDH     264                     // padded f16/f32 tile stride for 256 cols
#define LROWS   64                      // rows per LSTM block (4 M-tiles)

typedef __attribute__((ext_vector_type(16))) _Float16 v16h;
typedef __attribute__((ext_vector_type(8)))  _Float16 v8h;
typedef __attribute__((ext_vector_type(8)))  float    v8f;

// ---------------- WMMA helpers (CDNA5 16x16x32 f16 -> f32) ----------------
__device__ __forceinline__ v8f wmma_f16(v16h a, v16h b, v8f c) {
    return __builtin_amdgcn_wmma_f32_16x16x32_f16(
        /*neg_a=*/false, a, /*neg_b=*/false, b,
        /*c_mod=*/(short)0, c, /*reuse_a=*/false, /*reuse_b=*/false);
}

// A fragment: 16x32 f16, row-major source with stride lda (halfs).
__device__ __forceinline__ v16h load_a_frag(const _Float16* A, int lda, int kb, int lane) {
    int row = lane & 15, g = lane >> 4;
    const _Float16* p = A + row * lda + kb + g * 8;
    v8h lo = *(const v8h*)(p);
    v8h hi = *(const v8h*)(p + 16);
    return __builtin_shufflevector(lo, hi, 0,1,2,3,4,5,6,7,8,9,10,11,12,13,14,15);
}

// B fragment: 32x16 f16 from weight stored [N][K] (K contiguous, stride ldb halfs).
__device__ __forceinline__ v16h load_b_frag(const _Float16* Bw, int ldb, int n0, int kb, int lane) {
    int col = lane & 15, g = lane >> 4;
    const _Float16* p = Bw + (size_t)(n0 + col) * ldb + kb + g * 8;
    v8h lo = *(const v8h*)(p);
    v8h hi = *(const v8h*)(p + 16);
    return __builtin_shufflevector(lo, hi, 0,1,2,3,4,5,6,7,8,9,10,11,12,13,14,15);
}

// C fragment store: element (m = r + 8*(lane>=16), n = lane&15)
__device__ __forceinline__ void store_c_frag(float* C, int ldc, int n0, v8f c, int lane) {
    int n = lane & 15, g = lane >> 4;
#pragma unroll
    for (int r = 0; r < 8; ++r)
        C[(r + 8 * g) * ldc + n0 + n] = c[r];
}

// Async global -> LDS 16B copy (CDNA5 TDM-family path, tracked by ASYNCcnt).
// lds_off is the wave-relative LDS byte offset (= low 32 bits of generic ptr).
__device__ __forceinline__ void async_copy_b128(unsigned lds_off, const void* gptr) {
    asm volatile("global_load_async_to_lds_b128 %0, %1, off"
                 :: "v"(lds_off), "v"(gptr) : "memory");
}
__device__ __forceinline__ void wait_asynccnt0() {
    asm volatile("s_wait_asynccnt 0" ::: "memory");
}

// GEMM for M=64 (4 m-tiles) x N=256 (16 n-tiles): each wave owns 2 n-tiles and
// keeps 4 m-tile accumulators in registers so each B fragment feeds 4 WMMAs.
__device__ void gemm64x256(const _Float16* A, int lda, const _Float16* Bw, int ldb,
                           float* C, int ldc, const float* bias, int K, int tid) {
    const int lane = tid & 31, wave = tid >> 5;
    for (int nt = wave * 2; nt < wave * 2 + 2; ++nt) {
        const int n0 = nt * 16;
        v8f zero = {};
        v8f acc0 = zero, acc1 = zero, acc2 = zero, acc3 = zero;
        for (int kb = 0; kb < K; kb += 32) {
            v16h a0 = load_a_frag(A +  0 * lda, lda, kb, lane);
            v16h a1 = load_a_frag(A + 16 * lda, lda, kb, lane);
            v16h a2 = load_a_frag(A + 32 * lda, lda, kb, lane);
            v16h a3 = load_a_frag(A + 48 * lda, lda, kb, lane);
            v16h bf = load_b_frag(Bw, ldb, n0, kb, lane);
            acc0 = wmma_f16(a0, bf, acc0);
            acc1 = wmma_f16(a1, bf, acc1);
            acc2 = wmma_f16(a2, bf, acc2);
            acc3 = wmma_f16(a3, bf, acc3);
        }
        if (bias) {
            float bv = bias[n0 + (lane & 15)];
#pragma unroll
            for (int r = 0; r < 8; ++r) { acc0[r] += bv; acc1[r] += bv; acc2[r] += bv; acc3[r] += bv; }
        }
        store_c_frag(C +  0 * ldc, ldc, n0, acc0, lane);
        store_c_frag(C + 16 * ldc, ldc, n0, acc1, lane);
        store_c_frag(C + 32 * ldc, ldc, n0, acc2, lane);
        store_c_frag(C + 48 * ldc, ldc, n0, acc3, lane);
    }
}

__device__ __forceinline__ float sigmoidf_(float x) { return 1.0f / (1.0f + expf(-x)); }

// ---------------- weight prep kernels ----------------
__global__ void k_convert_copy(const float* __restrict__ src, _Float16* __restrict__ dst, int total) {
    int i = blockIdx.x * blockDim.x + threadIdx.x;
    if (i < total) dst[i] = (_Float16)src[i];
}

// src [K][N] row-major -> dst [Np][Kp], dst[n][k] = src[k][n], zero pad
__global__ void k_convert_transpose_pad(const float* __restrict__ src, _Float16* __restrict__ dst,
                                        int K, int N, int Kp, int Np) {
    int i = blockIdx.x * blockDim.x + threadIdx.x;
    if (i >= Np * Kp) return;
    int n = i / Kp, k = i % Kp;
    float v = (n < N && k < K) ? src[(size_t)k * N + n] : 0.0f;
    dst[i] = (_Float16)v;
}

// ---------------- stage 1: obs encoder + LSTM cell ----------------
__global__ __launch_bounds__(256)
void k_lstm(const float* __restrict__ obs, const float* __restrict__ lstm_h,
            const float* __restrict__ lstm_c, const float* __restrict__ masks,
            const float* __restrict__ b_obs, const float* __restrict__ b_ih,
            const float* __restrict__ b_hh,
            const _Float16* __restrict__ Wobs_h, const _Float16* __restrict__ Wih_h,
            const _Float16* __restrict__ Whh_h,
            float* __restrict__ h_out, float* __restrict__ c_out,
            _Float16* __restrict__ hid16_out /* optional f16 hidden side-channel */) {
    __shared__ __align__(16) _Float16 sh_obs[LROWS * OBSP];
    __shared__ __align__(16) _Float16 sh_x[LROWS * LDH];
    __shared__ __align__(16) _Float16 sh_hp[LROWS * LDH];

    const int tid = threadIdx.x;
    const size_t r0 = (size_t)blockIdx.x * LROWS;

    // stage obs (padded K: 100->128) and hp = h*mask into LDS as f16
    for (int i = tid; i < LROWS * OBSP; i += 256) {
        int m = i / OBSP, k = i % OBSP;
        float v = (k < OBS_) ? obs[(r0 + m) * OBS_ + k] : 0.0f;
        sh_obs[i] = (_Float16)v;
    }
    for (int i = tid; i < LROWS * HDIM; i += 256) {
        int m = i / HDIM, k = i % HDIM;
        size_t row = r0 + m;
        sh_hp[m * LDH + k] = (_Float16)(lstm_h[row * HDIM + k] * masks[row]);
    }
    // prefetch lstm_c rows used in the epilogue (gfx1250 global_prefetch)
    for (int i = tid; i < LROWS * 8; i += 256)
        __builtin_prefetch(lstm_c + r0 * HDIM + (size_t)i * 32, 0, 1);
    __syncthreads();

    const int lane = tid & 31, wave = tid >> 5;

    // x = obs @ W_obs + b_obs  (64 x 256): 2 n-tiles per wave, 4 m-tiles shared B
    for (int nt = wave * 2; nt < wave * 2 + 2; ++nt) {
        const int n0 = nt * 16;
        v8f zero = {};
        v8f acc[4] = {zero, zero, zero, zero};
#pragma unroll
        for (int kb = 0; kb < OBSP; kb += 32) {
            v16h a0 = load_a_frag(sh_obs +  0 * OBSP, OBSP, kb, lane);
            v16h a1 = load_a_frag(sh_obs + 16 * OBSP, OBSP, kb, lane);
            v16h a2 = load_a_frag(sh_obs + 32 * OBSP, OBSP, kb, lane);
            v16h a3 = load_a_frag(sh_obs + 48 * OBSP, OBSP, kb, lane);
            v16h bf = load_b_frag(Wobs_h, OBSP, n0, kb, lane);
            acc[0] = wmma_f16(a0, bf, acc[0]);
            acc[1] = wmma_f16(a1, bf, acc[1]);
            acc[2] = wmma_f16(a2, bf, acc[2]);
            acc[3] = wmma_f16(a3, bf, acc[3]);
        }
        float bv = b_obs[n0 + (lane & 15)];
        int n = lane & 15, g = lane >> 4;
#pragma unroll
        for (int mt = 0; mt < 4; ++mt)
#pragma unroll
            for (int r = 0; r < 8; ++r)
                sh_x[(mt * 16 + r + 8 * g) * LDH + n0 + n] = (_Float16)(acc[mt][r] + bv);
    }
    __syncthreads();

    // gates: each wave owns 2 n-tiles of H; 4 gates x 4 m-tiles in regs.
    for (int nt = wave * 2; nt < wave * 2 + 2; ++nt) {
        const int n0 = nt * 16;
        v8f zero = {};
        v8f acc[4][4];
#pragma unroll
        for (int gt = 0; gt < 4; ++gt)
#pragma unroll
            for (int mt = 0; mt < 4; ++mt) acc[gt][mt] = zero;

        for (int pass = 0; pass < 2; ++pass) {
            const _Float16* As = pass ? sh_hp : sh_x;
            const _Float16* Bw = pass ? Whh_h : Wih_h;
            for (int kb = 0; kb < HDIM; kb += 32) {
                v16h a0 = load_a_frag(As +  0 * LDH, LDH, kb, lane);
                v16h a1 = load_a_frag(As + 16 * LDH, LDH, kb, lane);
                v16h a2 = load_a_frag(As + 32 * LDH, LDH, kb, lane);
                v16h a3 = load_a_frag(As + 48 * LDH, LDH, kb, lane);
#pragma unroll
                for (int gt = 0; gt < 4; ++gt) {
                    v16h bf = load_b_frag(Bw, HDIM, gt * 256 + n0, kb, lane);
                    acc[gt][0] = wmma_f16(a0, bf, acc[gt][0]);
                    acc[gt][1] = wmma_f16(a1, bf, acc[gt][1]);
                    acc[gt][2] = wmma_f16(a2, bf, acc[gt][2]);
                    acc[gt][3] = wmma_f16(a3, bf, acc[gt][3]);
                }
            }
        }

        // fused LSTM epilogue directly on C fragments
        const int ncol = lane & 15, g = lane >> 4;
        const int col = n0 + ncol;
        float bi = b_ih[col]       + b_hh[col];
        float bf = b_ih[256 + col] + b_hh[256 + col];
        float bg = b_ih[512 + col] + b_hh[512 + col];
        float bo = b_ih[768 + col] + b_hh[768 + col];
#pragma unroll
        for (int mt = 0; mt < 4; ++mt) {
#pragma unroll
            for (int r = 0; r < 8; ++r) {
                size_t row = r0 + mt * 16 + r + 8 * g;
                float mk = masks[row];
                float cp = lstm_c[row * HDIM + col] * mk;
                float ig = acc[0][mt][r] + bi;
                float fg = acc[1][mt][r] + bf;
                float gv = acc[2][mt][r] + bg;
                float og = acc[3][mt][r] + bo;
                float cn = sigmoidf_(fg) * cp + sigmoidf_(ig) * tanhf(gv);
                float hn = sigmoidf_(og) * tanhf(cn);
                h_out[row * HDIM + col] = hn;
                c_out[row * HDIM + col] = cn;
                if (hid16_out) hid16_out[row * HDIM + col] = (_Float16)hn;
            }
        }
    }
}

// ---------------- stage 2: message encoder + GAT1 + GAT2 + decoder + logits ----
// 2 batches per block: rows 0..31 = batch A, rows 32..63 = batch B (4 M-tiles).
__global__ __launch_bounds__(256)
void k_gat(const float* __restrict__ h_src, const _Float16* __restrict__ hid16,
           const float* __restrict__ masks,
           const float* __restrict__ b_me,
           const float* __restrict__ a_i1, const float* __restrict__ a_j1,
           const float* __restrict__ a_i2, const float* __restrict__ a_j2,
           const float* __restrict__ b_md, const float* __restrict__ b_act,
           const _Float16* __restrict__ Wme_h, const _Float16* __restrict__ W1_h,
           const _Float16* __restrict__ W2_h, const _Float16* __restrict__ Wmd_h,
           const _Float16* __restrict__ Wact_h,
           float* __restrict__ logits) {
    __shared__ __align__(16) _Float16 sh_hid[64 * LDH];   // hidden f16
    __shared__ __align__(16) _Float16 sh_a[64 * LDH];     // reused A buffer
    __shared__ __align__(16) float    sh_c[64 * LDH];     // f32 GEMM output
    __shared__ float sh_att[2 * HEADS_ * NAG * 32];       // att1, reused for att2
    __shared__ float sh_ci[2 * HEADS_ * 32], sh_cj[2 * HEADS_ * 32];
    __shared__ float sh_ci2[64], sh_cj2[64], sh_mask[64];

    const int tid = threadIdx.x;
    const int b0 = blockIdx.x * 2;
    const int lane = tid & 31, wave = tid >> 5;

    // load hidden: async 16B global->LDS copies when f16 side-channel present
    if (hid16) {
        for (int t = tid; t < 2 * NAG * 32; t += 256) {   // 40 rows x 32 chunks of 16B
            int row = t >> 5, ch = t & 31;
            int bb = row / NAG, m = row % NAG;
            unsigned lds_off = (unsigned)(size_t)(&sh_hid[(bb * 32 + m) * LDH]) + (unsigned)(ch * 16);
            const _Float16* g = hid16 + ((size_t)(b0 + bb) * NAG + m) * HDIM + ch * 8;
            async_copy_b128(lds_off, g);
        }
    } else {
        for (int idx = tid; idx < 2 * NAG * HDIM; idx += 256) {
            int bb = idx / (NAG * HDIM), rem = idx % (NAG * HDIM);
            int m = rem / HDIM, k = rem % HDIM;
            size_t src = ((size_t)(b0 + bb) * NAG + m) * HDIM + k;
            sh_hid[(bb * 32 + m) * LDH + k] = (_Float16)h_src[src];
        }
    }
    // zero pad rows {20..31, 52..63} of f16 A buffers (overlaps with async copies)
    for (int i = tid; i < 24 * LDH; i += 256) {
        int mm = i / LDH, k = i % LDH;
        int row = (mm < 12) ? (20 + mm) : (40 + mm);
        sh_hid[row * LDH + k] = (_Float16)0.0f;
        sh_a[row * LDH + k]   = (_Float16)0.0f;
    }
    if (tid < 64) {
        int bb = tid >> 5, i = tid & 31;
        sh_mask[tid] = (i < NAG) ? masks[(size_t)(b0 + bb) * NAG + i] : 0.0f;
    }
    if (hid16) wait_asynccnt0();
    __syncthreads();

    // comm = hidden @ W_me + b_me
    gemm64x256(sh_hid, LDH, Wme_h, HDIM, sh_c, LDH, b_me, HDIM, tid);
    __syncthreads();
    for (int idx = tid; idx < 2 * NAG * HDIM; idx += 256) {
        int bb = idx / (NAG * HDIM), rem = idx % (NAG * HDIM);
        int m = bb * 32 + rem / HDIM, k = rem % HDIM;
        sh_a[m * LDH + k] = (_Float16)sh_c[m * LDH + k];
    }
    __syncthreads();

    // GAT1: h1 = comm @ W1
    gemm64x256(sh_a, LDH, W1_h, HDIM, sh_c, LDH, nullptr, HDIM, tid);
    __syncthreads();

    // ci/cj per (batch, head, node): 320 dot-64 tasks
    for (int t = tid; t < 320; t += 256) {
        int which = t / 160, u = t % 160;
        int bb = u / 80, v = u % 80, hd = v / NAG, i = v % NAG;
        const float* av = which ? a_j1 : a_i1;
        float s = 0.f;
        for (int d = 0; d < GH_; ++d)
            s += sh_c[(bb * 32 + i) * LDH + hd * GH_ + d] * av[hd * GH_ + d];
        (which ? sh_cj : sh_ci)[(bb * HEADS_ + hd) * 32 + i] = s;
    }
    __syncthreads();

    // masked softmax rows (leaky_relu 0.2, forced self-loops): 160 rows
    if (tid < 160) {
        int bb = tid / 80, t = tid % 80, hd = t / NAG, i = t % NAG;
        const float* mrow = &sh_mask[bb * 32];
        float ci = sh_ci[(bb * HEADS_ + hd) * 32 + i], mi = mrow[i];
        float e[NAG], adjv[NAG], mx = -3.4e38f;
        for (int j = 0; j < NAG; ++j) {
            float adj = (i == j) ? 1.0f : mi * mrow[j];
            float v;
            if (adj > 0.0f) { v = ci + sh_cj[(bb * HEADS_ + hd) * 32 + j]; v = (v < 0.0f) ? 0.2f * v : v; }
            else v = -9.0e15f;
            e[j] = v; adjv[j] = adj; mx = fmaxf(mx, v);
        }
        float sum = 0.f;
        for (int j = 0; j < NAG; ++j) { e[j] = expf(e[j] - mx); sum += e[j]; }
        float inv = 1.0f / sum;
        for (int j = 0; j < NAG; ++j)
            sh_att[((bb * HEADS_ + hd) * NAG + i) * 32 + j] = e[j] * inv * adjv[j];
    }
    __syncthreads();

    // out1 = att1 @ h1, ELU -> f16 (overwrite sh_a; comm is dead)
    for (int idx = tid; idx < 2 * NAG * HDIM; idx += 256) {
        int bb = idx / (NAG * HDIM), rem = idx % (NAG * HDIM);
        int i = rem >> 8, c = rem & 255, hd = c >> 6;
        const float* att = &sh_att[((bb * HEADS_ + hd) * NAG + i) * 32];
        float s = 0.f;
        for (int j = 0; j < NAG; ++j) s += att[j] * sh_c[(bb * 32 + j) * LDH + c];
        s = (s > 0.0f) ? s : (expf(s) - 1.0f);
        sh_a[(bb * 32 + i) * LDH + c] = (_Float16)s;
    }
    __syncthreads();

    // GAT2: h2 = elu_out @ W2
    gemm64x256(sh_a, LDH, W2_h, HDIM, sh_c, LDH, nullptr, HDIM, tid);
    __syncthreads();

    // ci2/cj2: 80 dot-256 tasks
    if (tid < 80) {
        int which = tid / 40, u = tid % 40, bb = u / NAG, i = u % NAG;
        const float* av = which ? a_j2 : a_i2;
        float s = 0.f;
        for (int d = 0; d < HDIM; ++d) s += sh_c[(bb * 32 + i) * LDH + d] * av[d];
        (which ? sh_cj2 : sh_ci2)[bb * 32 + i] = s;
    }
    __syncthreads();

    // att2 softmax: 40 rows (reuse front of sh_att)
    if (tid < 40) {
        int bb = tid / NAG, i = tid % NAG;
        const float* mrow = &sh_mask[bb * 32];
        float ci = sh_ci2[bb * 32 + i], mi = mrow[i];
        float e[NAG], adjv[NAG], mx = -3.4e38f;
        for (int j = 0; j < NAG; ++j) {
            float adj = (i == j) ? 1.0f : mi * mrow[j];
            float v;
            if (adj > 0.0f) { v = ci + sh_cj2[bb * 32 + j]; v = (v < 0.0f) ? 0.2f * v : v; }
            else v = -9.0e15f;
            e[j] = v; adjv[j] = adj; mx = fmaxf(mx, v);
        }
        float sum = 0.f;
        for (int j = 0; j < NAG; ++j) { e[j] = expf(e[j] - mx); sum += e[j]; }
        float inv = 1.0f / sum;
        for (int j = 0; j < NAG; ++j)
            sh_att[(bb * NAG + i) * 32 + j] = e[j] * inv * adjv[j];
    }
    __syncthreads();

    // out2 = att2 @ h2 -> f16 into sh_a
    for (int idx = tid; idx < 2 * NAG * HDIM; idx += 256) {
        int bb = idx / (NAG * HDIM), rem = idx % (NAG * HDIM);
        int i = rem >> 8, c = rem & 255;
        float s = 0.f;
        for (int j = 0; j < NAG; ++j)
            s += sh_att[(bb * NAG + i) * 32 + j] * sh_c[(bb * 32 + j) * LDH + c];
        sh_a[(bb * 32 + i) * LDH + c] = (_Float16)s;
    }
    __syncthreads();

    // commd = out2 @ W_md + b_md
    gemm64x256(sh_a, LDH, Wmd_h, HDIM, sh_c, LDH, b_md, HDIM, tid);
    __syncthreads();

    // masked commd -> f16 (reuse sh_a)
    for (int idx = tid; idx < 2 * NAG * HDIM; idx += 256) {
        int bb = idx / (NAG * HDIM), rem = idx % (NAG * HDIM);
        int m = bb * 32 + rem / HDIM, k = rem % HDIM;
        sh_a[m * LDH + k] = (_Float16)(sh_c[m * LDH + k] * sh_mask[m]);
    }
    __syncthreads();

    // logits = [hidden | commd] @ W_act : split-K (256+256), 4 M-tiles on waves 0..3
    if (wave < 4) {
        int m0 = wave * 16;
        v8f acc = {};
#pragma unroll
        for (int kb = 0; kb < HDIM; kb += 32) {
            v16h a  = load_a_frag(sh_hid + m0 * LDH, LDH, kb, lane);
            v16h bf = load_b_frag(Wact_h, 512, 0, kb, lane);
            acc = wmma_f16(a, bf, acc);
        }
#pragma unroll
        for (int kb = 0; kb < HDIM; kb += 32) {
            v16h a  = load_a_frag(sh_a + m0 * LDH, LDH, kb, lane);
            v16h bf = load_b_frag(Wact_h, 512, 0, kb + 256, lane);
            acc = wmma_f16(a, bf, acc);
        }
        store_c_frag(sh_c + m0 * LDH, LDH, 0, acc, lane);
    }
    __syncthreads();

    for (int idx = tid; idx < 2 * NAG * ACT_; idx += 256) {
        int bb = idx / (NAG * ACT_), rem = idx % (NAG * ACT_);
        int m = rem / ACT_, c = rem % ACT_;
        logits[((size_t)(b0 + bb) * NAG + m) * ACT_ + c] = sh_c[(bb * 32 + m) * LDH + c] + b_act[c];
    }
}

// ---------------- launcher ----------------
extern "C" void kernel_launch(void* const* d_in, const int* in_sizes, int n_in,
                              void* d_out, int out_size, void* d_ws, size_t ws_size,
                              hipStream_t stream) {
    const float* obs    = (const float*)d_in[0];
    const float* lstm_h = (const float*)d_in[1];
    const float* lstm_c = (const float*)d_in[2];
    const float* masks  = (const float*)d_in[3];
    const float* W_obs  = (const float*)d_in[4];
    const float* b_obs  = (const float*)d_in[5];
    const float* W_ih   = (const float*)d_in[6];
    const float* W_hh   = (const float*)d_in[7];
    const float* b_ih   = (const float*)d_in[8];
    const float* b_hh   = (const float*)d_in[9];
    const float* W_me   = (const float*)d_in[10];
    const float* b_me   = (const float*)d_in[11];
    const float* W1     = (const float*)d_in[12];
    const float* a_i1   = (const float*)d_in[13];
    const float* a_j1   = (const float*)d_in[14];
    const float* W2     = (const float*)d_in[15];
    const float* a_i2   = (const float*)d_in[16];
    const float* a_j2   = (const float*)d_in[17];
    const float* W_md   = (const float*)d_in[18];
    const float* b_md   = (const float*)d_in[19];
    const float* W_act  = (const float*)d_in[20];
    const float* b_act  = (const float*)d_in[21];

    // output layout: logits | h_new | c_new
    float* out    = (float*)d_out;
    float* logits = out;
    float* h_out  = out + (size_t)BN * ACT_;
    float* c_out  = h_out + (size_t)BN * HDIM;

    // workspace: f16 weights ([N][K], K contiguous), then optional f16 hidden buffer
    _Float16* Wobs_h = (_Float16*)d_ws;               // 256 x 128
    _Float16* Wih_h  = Wobs_h + 256 * OBSP;           // 1024 x 256
    _Float16* Whh_h  = Wih_h  + G4H * HDIM;           // 1024 x 256
    _Float16* Wme_h  = Whh_h  + G4H * HDIM;           // 256 x 256
    _Float16* W1_h   = Wme_h  + HDIM * HDIM;
    _Float16* W2_h   = W1_h   + HDIM * HDIM;
    _Float16* Wmd_h  = W2_h   + HDIM * HDIM;
    _Float16* Wact_h = Wmd_h  + HDIM * HDIM;          // 16 x 512

    size_t weight_halfs = 256 * OBSP + 2 * (size_t)G4H * HDIM + 4 * (size_t)HDIM * HDIM + 16 * 512;
    size_t need16 = weight_halfs * sizeof(_Float16) + (size_t)BN * HDIM * sizeof(_Float16);
    _Float16* hid16 = (ws_size >= need16) ? (Wact_h + 16 * 512) : nullptr;

    // ---- weight prep ----
    k_convert_transpose_pad<<<(256 * OBSP + 255) / 256, 256, 0, stream>>>(W_obs, Wobs_h, OBS_, HDIM, OBSP, HDIM);
    k_convert_copy<<<(G4H * HDIM + 255) / 256, 256, 0, stream>>>(W_ih, Wih_h, G4H * HDIM);
    k_convert_copy<<<(G4H * HDIM + 255) / 256, 256, 0, stream>>>(W_hh, Whh_h, G4H * HDIM);
    k_convert_transpose_pad<<<(HDIM * HDIM + 255) / 256, 256, 0, stream>>>(W_me, Wme_h, HDIM, HDIM, HDIM, HDIM);
    k_convert_transpose_pad<<<(HDIM * HDIM + 255) / 256, 256, 0, stream>>>(W1, W1_h, HDIM, HDIM, HDIM, HDIM);
    k_convert_transpose_pad<<<(HDIM * HDIM + 255) / 256, 256, 0, stream>>>(W2, W2_h, HDIM, HDIM, HDIM, HDIM);
    k_convert_transpose_pad<<<(HDIM * HDIM + 255) / 256, 256, 0, stream>>>(W_md, Wmd_h, HDIM, HDIM, HDIM, HDIM);
    k_convert_transpose_pad<<<(16 * 512 + 255) / 256, 256, 0, stream>>>(W_act, Wact_h, 2 * HDIM, ACT_, 2 * HDIM, 16);

    // ---- stage 1: encoder + LSTM (64 rows/block) ----
    k_lstm<<<BN / LROWS, 256, 0, stream>>>(obs, lstm_h, lstm_c, masks, b_obs, b_ih, b_hh,
                                           Wobs_h, Wih_h, Whh_h, h_out, c_out, hid16);

    // ---- stage 2: GAT comm + logits (2 batches/block) ----
    k_gat<<<BATCH / 2, 256, 0, stream>>>(h_out, hid16, masks, b_me, a_i1, a_j1, a_i2, a_j2,
                                         b_md, b_act, Wme_h, W1_h, W2_h, Wmd_h, Wact_h, logits);
}
